// RNN_Layer_61194694033430
// MI455X (gfx1250) — matrix operand order repaired and checked by reading, not compiled, not run on hardware
//
#include <hip/hip_runtime.h>
#include <math.h>

typedef _Float16 v8h  __attribute__((ext_vector_type(8)));
typedef _Float16 v16h __attribute__((ext_vector_type(16)));
typedef float    v8f  __attribute__((ext_vector_type(8)));

constexpr int Bsz = 64;
constexpr int Tsz = 512;
constexpr int Dd  = 256;
constexpr int Hh  = 256;
constexpr long BT = (long)Bsz * Tsz;               // 32768
constexpr long LS_OFF = (long)Bsz * Tsz * 2 * Hh;  // start of last_state in d_out

// Hardware tanh (V_TANH_F32) when the builtin exists; libm fallback otherwise.
#if __has_builtin(__builtin_amdgcn_tanhf)
#define FAST_TANH(x) __builtin_amdgcn_tanhf(x)
#elif __has_builtin(__builtin_amdgcn_tanh_f32)
#define FAST_TANH(x) __builtin_amdgcn_tanh_f32(x)
#else
#define FAST_TANH(x) tanhf(x)
#endif

static __device__ inline v16h cat8(v8h lo, v8h hi) {
    return __builtin_shufflevector(lo, hi, 0,1,2,3,4,5,6,7,8,9,10,11,12,13,14,15);
}

// ---------------------------------------------------------------------------
// Pack: inputs f32 -> f16 (K-major already); W_ih/W_hh f32 -> f16 transposed
// (K-major per output column => WMMA B operand is one contiguous 32B load);
// mask [B][T] -> [T][B] to match the step-major scan access pattern.
// ---------------------------------------------------------------------------
__global__ void pack_kernel(const float* __restrict__ in,
                            const float* __restrict__ wih,
                            const float* __restrict__ whh,
                            const float* __restrict__ mask,
                            _Float16* __restrict__ in16,
                            _Float16* __restrict__ wih_t,
                            _Float16* __restrict__ whh_t,
                            float* __restrict__ mask_t) {
    long idx = (long)blockIdx.x * blockDim.x + threadIdx.x;
    const long N0 = BT * Dd;
    const long N1 = N0 + (long)Dd * Hh;
    const long N2 = N1 + (long)Hh * Hh;
    const long N3 = N2 + BT;
    if (idx < N0) {
        in16[idx] = (_Float16)in[idx];
    } else if (idx < N1) {
        long j = idx - N0;
        int d = (int)(j / Hh), h = (int)(j % Hh);
        wih_t[(long)h * Dd + d] = (_Float16)wih[j];   // [h][d], K-major
    } else if (idx < N2) {
        long j = idx - N1;
        int k = (int)(j / Hh), n = (int)(j % Hh);
        whh_t[(long)n * Hh + k] = (_Float16)whh[j];   // [n][k], K-major
    } else if (idx < N3) {
        long j = idx - N2;
        int b = (int)(j / Tsz), t = (int)(j % Tsz);
        mask_t[(long)t * Bsz + b] = mask[j];          // [t][b]
    }
}

// ---------------------------------------------------------------------------
// x_proj = inputs @ W_ih + b  (M=32768, N=256, K=256), f32 accumulate.
// Block = 8 waves; each wave owns one 16-row M-tile (A held in 64 VGPRs)
// and sweeps all 16 N-tiles (B tiles are block-shared, served from cache).
// Output written step-major: xp[t][b][h].
// ---------------------------------------------------------------------------
__global__ __launch_bounds__(256) void
xproj_kernel(const _Float16* __restrict__ in16,
             const _Float16* __restrict__ wih_t,
             const float* __restrict__ bias,
             float* __restrict__ xp) {
    const int wave = threadIdx.x >> 5;          // 0..7
    const int lane = threadIdx.x & 31;
    const int s = lane >> 4, c = lane & 15;
    const int mtile = blockIdx.x * 8 + wave;    // 0..2047
    const long rowA = (long)(mtile * 16 + c) * Dd;

    v16h A[8];
#pragma unroll
    for (int kg = 0; kg < 8; ++kg) {
        v8h lo = *(const v8h*)(in16 + rowA + kg * 32 + s * 16);
        v8h hi = *(const v8h*)(in16 + rowA + kg * 32 + s * 16 + 8);
        A[kg] = cat8(lo, hi);
    }
    for (int nt = 0; nt < 16; ++nt) {
        const int n = nt * 16 + c;
        v8f acc = {0.f,0.f,0.f,0.f,0.f,0.f,0.f,0.f};
#pragma unroll
        for (int kg = 0; kg < 8; ++kg) {
            v16h Bk = *(const v16h*)(wih_t + (long)n * Dd + kg * 32 + s * 16);
            acc = __builtin_amdgcn_wmma_f32_16x16x32_f16(
                      false, A[kg], false, Bk, (short)0, acc, false, false);
        }
        const float bv = bias[n];
#pragma unroll
        for (int r = 0; r < 8; ++r) {
            const int m = mtile * 16 + r + 8 * s;   // row in [0, B*T)
            const int b = m >> 9;                   // m / T
            const int t = m & (Tsz - 1);            // m % T
            xp[((long)t * Bsz + b) * Hh + n] = acc[r] + bv;
        }
    }
}

// ---------------------------------------------------------------------------
// Persistent scan: 8 blocks x 512 threads (16 waves).
// blocks 0-3: forward, batch tiles 0-3; blocks 4-7: backward.
// Wave w owns output columns [16w,16w+16). W_hh B-tiles live in 64 VGPRs for
// the whole kernel; carried h is f16 in LDS, double-buffered, one barrier per
// step; next step's xp/mask are prefetched into registers before the barrier.
// ---------------------------------------------------------------------------
__global__ __launch_bounds__(512) void
scan_kernel(const _Float16* __restrict__ whh_t,
            const float* __restrict__ xq,      // [T][B][H]
            const float* __restrict__ mk,      // [T][B]
            float* __restrict__ out) {
    const int dir   = blockIdx.x >> 2;     // 0 = fwd, 1 = bwd
    const int btile = blockIdx.x & 3;
    const int wave  = threadIdx.x >> 5;    // 0..15 = n-tile
    const int lane  = threadIdx.x & 31;
    const int s = lane >> 4, c = lane & 15;
    const int n  = wave * 16 + c;
    const int b0 = btile * 16;

    __shared__ __align__(32) _Float16 hA[2][16][Hh];

    // W_hh B operand, register-resident for all 512 steps (64 VGPRs).
    v16h wB[8];
#pragma unroll
    for (int kg = 0; kg < 8; ++kg)
        wB[kg] = *(const v16h*)(whh_t + (long)n * Hh + kg * 32 + s * 16);

    // h0 = 0
    for (int i = threadIdx.x; i < 16 * Hh / 2; i += blockDim.x)
        ((unsigned int*)&hA[0][0][0])[i] = 0u;

    // prefetch xp/mask for step 0
    float xv[8], mv[8];
    {
        const int t0 = dir ? (Tsz - 1) : 0;
#pragma unroll
        for (int r = 0; r < 8; ++r) {
            const int b = b0 + r + 8 * s;
            xv[r] = xq[((long)t0 * Bsz + b) * Hh + n];
            mv[r] = dir ? mk[(long)t0 * Bsz + b] : 1.f;
        }
    }
    __syncthreads();

    int p = 0;
    for (int step = 0; step < Tsz; ++step) {
        const int t = dir ? (Tsz - 1 - step) : step;

        v8f acc = {0.f,0.f,0.f,0.f,0.f,0.f,0.f,0.f};
#pragma unroll
        for (int kg = 0; kg < 8; ++kg) {
            v8h lo = *(const v8h*)&hA[p][c][kg * 32 + s * 16];
            v8h hi = *(const v8h*)&hA[p][c][kg * 32 + s * 16 + 8];
            acc = __builtin_amdgcn_wmma_f32_16x16x32_f16(
                      false, cat8(lo, hi), false, wB[kg], (short)0, acc, false, false);
        }

        float hn[8];
#pragma unroll
        for (int r = 0; r < 8; ++r) {
            const int m = r + 8 * s;
            hn[r] = FAST_TANH(acc[r] + xv[r]);
            // carried state: fwd = hn, bwd = hn * mask (h0 == 0)
            const float carry = dir ? hn[r] * mv[r] : hn[r];
            hA[p ^ 1][m][n] = (_Float16)carry;
        }

        // global outputs (off the barrier-critical path)
#pragma unroll
        for (int r = 0; r < 8; ++r) {
            const int b = b0 + r + 8 * s;
            const long bt = (long)b * Tsz + t;
            out[bt * (2 * Hh) + dir * Hh + n] = hn[r];  // sequence (unmasked hn)
            // last_state: fwd half at final step; bwd half = hn of the first
            // reverse step (t = T-1), matching seq_b_rev[0] in the reference.
            if (dir == 0 && step == Tsz - 1) out[LS_OFF + (long)b * (2 * Hh) + n] = hn[r];
            if (dir == 1 && step == 0)       out[LS_OFF + (long)b * (2 * Hh) + Hh + n] = hn[r];
        }

        // prefetch next step's xp/mask; overlaps barrier + next WMMA phase
        if (step + 1 < Tsz) {
            const int tn = dir ? (Tsz - 2 - step) : (step + 1);
#pragma unroll
            for (int r = 0; r < 8; ++r) {
                const int b = b0 + r + 8 * s;
                xv[r] = xq[((long)tn * Bsz + b) * Hh + n];
                if (dir) mv[r] = mk[(long)tn * Bsz + b];
            }
        }
        __syncthreads();
        p ^= 1;
    }
}

// ---------------------------------------------------------------------------
extern "C" void kernel_launch(void* const* d_in, const int* in_sizes, int n_in,
                              void* d_out, int out_size, void* d_ws, size_t ws_size,
                              hipStream_t stream) {
    const float* inputs = (const float*)d_in[0];   // [B,T,D]
    const float* mask   = (const float*)d_in[1];   // [B,T]
    const float* W_ih   = (const float*)d_in[2];   // [D,H]
    const float* W_hh   = (const float*)d_in[3];   // [H,H]
    const float* bias   = (const float*)d_in[4];   // [H]
    float* out = (float*)d_out;

    // workspace carve-up (all offsets 32B aligned)
    char* ws = (char*)d_ws;
    size_t off = 0;
    float*    xp    = (float*)(ws + off);  off += (size_t)BT * Hh * 4;   // [T][B][H] f32
    _Float16* in16  = (_Float16*)(ws + off); off += (size_t)BT * Dd * 2; // [B*T][D] f16
    _Float16* wih_t = (_Float16*)(ws + off); off += (size_t)Dd * Hh * 2; // [H][D] f16
    _Float16* whh_t = (_Float16*)(ws + off); off += (size_t)Hh * Hh * 2; // [N][K] f16
    float*    mask_t= (float*)(ws + off);                                // [T][B] f32

    const long npack = BT * Dd + 2L * Dd * Hh + BT;
    pack_kernel<<<(int)((npack + 255) / 256), 256, 0, stream>>>(
        inputs, W_ih, W_hh, mask, in16, wih_t, whh_t, mask_t);

    xproj_kernel<<<(int)(BT / 16 / 8), 256, 0, stream>>>(in16, wih_t, bias, xp);

    scan_kernel<<<8, 512, 0, stream>>>(whh_t, xp, mask_t, out);
}
